// SummaryEncoder_72739566125262
// MI455X (gfx1250) — compile-verified, matrix operand order checked by reading, not run on hardware
//
#include <hip/hip_runtime.h>

#define B_  8
#define S_  512
#define J_  16
#define H_  768
#define L_  6
#define NH_ 12
#define DH_ 64
#define FF_ 3072
#define TOK_ (B_*S_)

typedef _Float16 f16;
typedef _Float16 v8h  __attribute__((ext_vector_type(8)));
typedef _Float16 v16h __attribute__((ext_vector_type(16)));
typedef float    v8f  __attribute__((ext_vector_type(8)));

// 16-byte vector type matching the async-to-LDS builtin's pointee
// ('__attribute__((__vector_size__(4*sizeof(int)))) int').
typedef int v4i __attribute__((vector_size(16)));
typedef __attribute__((address_space(1))) v4i as1_v4i;  // global (__device__)
typedef __attribute__((address_space(3))) v4i as3_v4i;  // LDS (__shared__)

#if defined(__has_builtin)
#if __has_builtin(__builtin_amdgcn_global_load_async_to_lds_b128)
#define USE_ASYNC 1
#endif
#endif
#ifndef USE_ASYNC
#define USE_ASYNC 0
#endif

static __device__ __forceinline__ v16h mk16(v8h a, v8h b) {
  v16h r;
#pragma unroll
  for (int i = 0; i < 8; ++i) { r[i] = a[i]; r[i + 8] = b[i]; }
  return r;
}

// D = A(16x32 f16) * B(32x16 f16) + C(16x16 f32)
static __device__ __forceinline__ v8f wmma16(v16h a, v16h b, v8f c) {
  return __builtin_amdgcn_wmma_f32_16x16x32_f16(false, a, false, b, (short)0, c,
                                                false, false);
}

// 16-byte global -> LDS copy.  Async (ASYNCcnt) when the gfx1250 builtin is
// available, otherwise a plain register round-trip.
static __device__ __forceinline__ void copy_b128(const f16* gsrc, f16* ldst) {
#if USE_ASYNC
  __builtin_amdgcn_global_load_async_to_lds_b128(
      (as1_v4i*)(__UINTPTR_TYPE__)gsrc,
      (as3_v4i*)(__UINTPTR_TYPE__)ldst, 0, 0);
#else
  *(v8h*)ldst = *(const v8h*)gsrc;
#endif
}

static __device__ __forceinline__ void wait_copies() {
#if USE_ASYNC
#if __has_builtin(__builtin_amdgcn_s_wait_asynccnt)
  __builtin_amdgcn_s_wait_asynccnt(0);
#else
  asm volatile("s_wait_asynccnt 0x0" ::: "memory");
#endif
#endif
}

// ---------------------------------------------------------------------------
// Embedding gather + LayerNorm.  One wave per token (wave32), 24 cols/lane.
// ---------------------------------------------------------------------------
__global__ __launch_bounds__(32)
void embed_ln_kernel(const int* __restrict__ ids, const int* __restrict__ segs,
                     const float* __restrict__ wemb, const float* __restrict__ pemb,
                     const float* __restrict__ temb, const float* __restrict__ lns,
                     const float* __restrict__ lnb,
                     float* __restrict__ h, f16* __restrict__ h16) {
  const int tok  = blockIdx.x;
  const int s    = tok % S_;
  const int lane = threadIdx.x;
  const int id   = ids[tok];
  const int sg   = segs[tok];
  const float* w = wemb + (size_t)id * H_;
  const float* p = pemb + (size_t)s * H_;
  const float* t = temb + (size_t)sg * H_;

  float v[24];
  float sum = 0.f;
#pragma unroll
  for (int i = 0; i < 24; ++i) {
    int c = lane + 32 * i;
    float x = w[c] + p[c] + t[c];
    v[i] = x; sum += x;
  }
#pragma unroll
  for (int m = 1; m < 32; m <<= 1) sum += __shfl_xor(sum, m, 32);
  const float mean = sum * (1.f / H_);
  float var = 0.f;
#pragma unroll
  for (int i = 0; i < 24; ++i) { float d = v[i] - mean; var += d * d; }
#pragma unroll
  for (int m = 1; m < 32; m <<= 1) var += __shfl_xor(var, m, 32);
  const float rstd = rsqrtf(var * (1.f / H_) + 1e-12f);
#pragma unroll
  for (int i = 0; i < 24; ++i) {
    int c = lane + 32 * i;
    float y = (v[i] - mean) * rstd * lns[c] + lnb[c];
    h  [(size_t)tok * H_ + c] = y;
    h16[(size_t)tok * H_ + c] = (f16)y;
  }
}

// ---------------------------------------------------------------------------
// Residual add + LayerNorm (in-place safe: per-token read-before-write).
// ---------------------------------------------------------------------------
__global__ __launch_bounds__(32)
void residual_ln_kernel(const float* __restrict__ hin, const float* __restrict__ delta,
                        const float* __restrict__ lns, const float* __restrict__ lnb,
                        float* __restrict__ hout, f16* __restrict__ h16) {
  const int tok  = blockIdx.x;
  const int lane = threadIdx.x;
  float v[24];
  float sum = 0.f;
#pragma unroll
  for (int i = 0; i < 24; ++i) {
    int c = lane + 32 * i;
    float x = hin[(size_t)tok * H_ + c] + delta[(size_t)tok * H_ + c];
    v[i] = x; sum += x;
  }
#pragma unroll
  for (int m = 1; m < 32; m <<= 1) sum += __shfl_xor(sum, m, 32);
  const float mean = sum * (1.f / H_);
  float var = 0.f;
#pragma unroll
  for (int i = 0; i < 24; ++i) { float d = v[i] - mean; var += d * d; }
#pragma unroll
  for (int m = 1; m < 32; m <<= 1) var += __shfl_xor(var, m, 32);
  const float rstd = rsqrtf(var * (1.f / H_) + 1e-12f);
#pragma unroll
  for (int i = 0; i < 24; ++i) {
    int c = lane + 32 * i;
    float y = (v[i] - mean) * rstd * lns[c] + lnb[c];
    hout[(size_t)tok * H_ + c] = y;
    h16 [(size_t)tok * H_ + c] = (f16)y;
  }
}

// ---------------------------------------------------------------------------
// Attention bias matrix: bias[b,q,k] = (1 - causal(q,k)*pad[b,q]) * -1e4
// ---------------------------------------------------------------------------
__global__ __launch_bounds__(512)
void build_bias_kernel(const int* __restrict__ start, const int* __restrict__ end,
                       const float* __restrict__ pad, float* __restrict__ bias) {
  const int q = blockIdx.x;
  const int b = blockIdx.y;
  const int k = threadIdx.x;
  unsigned mq = 0u, mk = 0u;
#pragma unroll
  for (int j = 0; j < J_; ++j) {
    int st = start[b * J_ + j];
    int en = end  [b * J_ + j];
    unsigned valid = !((st == 0) && (en == 0));
    if (valid && q >= st && q <= en) mq |= (1u << j);
    if (valid && k >= st && k <= en) mk |= (1u << j);
  }
  float causal = ((mq & mk) != 0u) ? 1.f : 0.f;
  float attn = causal * pad[b * S_ + q];
  bias[((size_t)b * S_ + q) * S_ + k] = (1.f - attn) * -10000.f;
}

// ---------------------------------------------------------------------------
// Weight convert+transpose: W (K x N, f32 row-major) -> Wt (N x K, f16).
// ---------------------------------------------------------------------------
__global__ __launch_bounds__(256)
void wtrans_kernel(const float* __restrict__ W, f16* __restrict__ Wt, int K, int N) {
  int idx = blockIdx.x * 256 + threadIdx.x;
  if (idx >= K * N) return;
  int n = idx % N;
  int k = idx / N;
  Wt[(size_t)n * K + k] = (f16)W[idx];
}

// ---------------------------------------------------------------------------
// V transpose: v16 (B*S, H) f16 -> vt (B, NH, DH, S) f16  (d-major per head)
// ---------------------------------------------------------------------------
__global__ __launch_bounds__(256)
void vtrans_kernel(const f16* __restrict__ v16, f16* __restrict__ vt) {
  int idx = blockIdx.x * 256 + threadIdx.x;
  if (idx >= TOK_ * H_) return;
  int c  = idx % H_;
  int sg = idx / H_;
  int b  = sg / S_;
  int s  = sg % S_;
  int hh = c / DH_;
  int d  = c % DH_;
  vt[(((size_t)b * NH_ + hh) * DH_ + d) * S_ + s] = v16[idx];
}

// ---------------------------------------------------------------------------
// WMMA GEMM:  C(MxN f32) = A(MxK f16, row-major) * Bt(NxK f16)^T + bias[N]
// Block: 256 threads = 8 waves; tile 128x128; wave => 32 rows x 64 cols
// (2 A-frags x 4 B-frags -> 8 WMMAs per 32-wide k-step).  Global->LDS is
// double-buffered through the gfx1250 async-to-LDS path (ASYNCcnt).
// act: 0 = none, 1 = exact GELU.  Cf and/or Ch may be null.
// ---------------------------------------------------------------------------
__global__ __launch_bounds__(256)
void gemm_wmma_kernel(const f16* __restrict__ A, const f16* __restrict__ Bt,
                      const float* __restrict__ bias,
                      float* __restrict__ Cf, f16* __restrict__ Ch,
                      int M, int N, int K, int act) {
  __shared__ __align__(16) f16 As[2][128][40];
  __shared__ __align__(16) f16 Bs[2][128][40];

  const int tid  = threadIdx.x;
  const int bm   = blockIdx.x * 128;
  const int bn   = blockIdx.y * 128;
  const int wave = tid >> 5;
  const int lane = tid & 31;
  const int lo   = lane & 15;
  const int hi   = lane >> 4;
  const int wm   = (wave & 3) * 32;   // wave M offset inside tile
  const int wn   = (wave >> 2) * 64;  // wave N offset inside tile

  // cooperative load: 512 b128-chunks per 128x32 tile; 2 per thread per array
  const int ra = tid >> 2;          // 0..63
  const int ca = (tid & 3) * 8;     // 0,8,16,24

  auto load_tile = [&](int k0, int buf) {
    copy_b128(&A [(size_t)(bm + ra)      * K + k0 + ca], &As[buf][ra][ca]);
    copy_b128(&A [(size_t)(bm + ra + 64) * K + k0 + ca], &As[buf][ra + 64][ca]);
    copy_b128(&Bt[(size_t)(bn + ra)      * K + k0 + ca], &Bs[buf][ra][ca]);
    copy_b128(&Bt[(size_t)(bn + ra + 64) * K + k0 + ca], &Bs[buf][ra + 64][ca]);
  };

  v8f acc[2][4];
#pragma unroll
  for (int i = 0; i < 2; ++i)
#pragma unroll
    for (int j = 0; j < 4; ++j) acc[i][j] = (v8f){};

  load_tile(0, 0);
  wait_copies();
  __syncthreads();

  const int nk = K >> 5;
  for (int kk = 0; kk < nk; ++kk) {
    const int cur = kk & 1;
    if (kk + 1 < nk) load_tile((kk + 1) * 32, cur ^ 1);

    v16h af[2], bf[4];
#pragma unroll
    for (int i = 0; i < 2; ++i) {
      const f16* p = &As[cur][wm + i * 16 + lo][0];
      af[i] = mk16(*(const v8h*)(p + hi * 8), *(const v8h*)(p + 16 + hi * 8));
    }
#pragma unroll
    for (int j = 0; j < 4; ++j) {
      const f16* p = &Bs[cur][wn + j * 16 + lo][0];
      bf[j] = mk16(*(const v8h*)(p + hi * 8), *(const v8h*)(p + 16 + hi * 8));
    }
#pragma unroll
    for (int i = 0; i < 2; ++i)
#pragma unroll
      for (int j = 0; j < 4; ++j)
        acc[i][j] = wmma16(af[i], bf[j], acc[i][j]);

    wait_copies();
    __syncthreads();
  }

#pragma unroll
  for (int i = 0; i < 2; ++i) {
#pragma unroll
    for (int r = 0; r < 8; ++r) {
      int row = bm + wm + i * 16 + r + 8 * hi;
#pragma unroll
      for (int j = 0; j < 4; ++j) {
        int col = bn + wn + j * 16 + lo;
        float x = acc[i][j][r] + bias[col];
        if (act == 1) x = 0.5f * x * (1.0f + erff(x * 0.70710678118f));
        if (Cf) Cf[(size_t)row * N + col] = x;
        if (Ch) Ch[(size_t)row * N + col] = (f16)x;
      }
    }
  }
}

// ---------------------------------------------------------------------------
// Flash-style attention.  One wave per (b, head, 16-query tile).
// ---------------------------------------------------------------------------
__global__ __launch_bounds__(128)
void attention_kernel(const f16* __restrict__ q16, const f16* __restrict__ k16,
                      const f16* __restrict__ vt, const float* __restrict__ bias,
                      f16* __restrict__ ctx16) {
  __shared__ __align__(16) f16 P[4][16][40];

  const int wid  = threadIdx.x >> 5;
  const int lane = threadIdx.x & 31;
  const int lo   = lane & 15;
  const int hi   = lane >> 4;

  const int gw   = blockIdx.x * 4 + wid;   // 0..3071
  const int qt   = gw & 31;                // query tile
  const int hb   = gw >> 5;
  const int head = hb % NH_;
  const int b    = hb / NH_;
  const int qrowbase = b * S_ + qt * 16;

  v16h qf[2];
#pragma unroll
  for (int ks = 0; ks < 2; ++ks) {
    const f16* qp = q16 + (size_t)(qrowbase + lo) * H_ + head * DH_ + ks * 32;
    qf[ks] = mk16(*(const v8h*)(qp + hi * 8), *(const v8h*)(qp + 16 + hi * 8));
  }

  v8f acc[4] = {{}, {}, {}, {}};
  float mrun[8], lrun[8];
#pragma unroll
  for (int r = 0; r < 8; ++r) { mrun[r] = -3.0e38f; lrun[r] = 0.f; }

  const float scale = 0.125f;  // 1/sqrt(64)
  const float* bp = bias + (size_t)b * S_ * S_;

  for (int kt = 0; kt < S_; kt += 32) {
    v8f s0 = {}, s1 = {};
#pragma unroll
    for (int ks = 0; ks < 2; ++ks) {
      const f16* kp0 = k16 + (size_t)(b * S_ + kt + lo) * H_ + head * DH_ + ks * 32;
      const f16* kp1 = kp0 + (size_t)16 * H_;
      v16h bf0 = mk16(*(const v8h*)(kp0 + hi * 8), *(const v8h*)(kp0 + 16 + hi * 8));
      v16h bf1 = mk16(*(const v8h*)(kp1 + hi * 8), *(const v8h*)(kp1 + 16 + hi * 8));
      s0 = wmma16(qf[ks], bf0, s0);
      s1 = wmma16(qf[ks], bf1, s1);
    }

    float p0[8], p1[8];
#pragma unroll
    for (int r = 0; r < 8; ++r) {
      int qr = qt * 16 + r + 8 * hi;
      float x0 = s0[r] * scale + bp[(size_t)qr * S_ + kt + lo];
      float x1 = s1[r] * scale + bp[(size_t)qr * S_ + kt + 16 + lo];
      float tmax = fmaxf(x0, x1);
#pragma unroll
      for (int m = 1; m < 16; m <<= 1) tmax = fmaxf(tmax, __shfl_xor(tmax, m, 32));
      float mnew = fmaxf(mrun[r], tmax);
      float sc   = __expf(mrun[r] - mnew);
      p0[r] = __expf(x0 - mnew);
      p1[r] = __expf(x1 - mnew);
      float rs = p0[r] + p1[r];
#pragma unroll
      for (int m = 1; m < 16; m <<= 1) rs += __shfl_xor(rs, m, 32);
      lrun[r] = lrun[r] * sc + rs;
      mrun[r] = mnew;
#pragma unroll
      for (int n = 0; n < 4; ++n) acc[n][r] *= sc;
    }

    __syncthreads();
#pragma unroll
    for (int r = 0; r < 8; ++r) {
      P[wid][r + 8 * hi][lo]      = (f16)p0[r];
      P[wid][r + 8 * hi][16 + lo] = (f16)p1[r];
    }
    __syncthreads();

    const f16* pp = &P[wid][lo][0];
    v16h pf = mk16(*(const v8h*)(pp + hi * 8), *(const v8h*)(pp + 16 + hi * 8));

#pragma unroll
    for (int n = 0; n < 4; ++n) {
      const f16* vp = vt + (((size_t)b * NH_ + head) * DH_ + n * 16 + lo) * S_ + kt;
      v16h vf = mk16(*(const v8h*)(vp + hi * 8), *(const v8h*)(vp + 16 + hi * 8));
      acc[n] = wmma16(pf, vf, acc[n]);
    }
    __syncthreads();
  }

#pragma unroll
  for (int r = 0; r < 8; ++r) {
    float inv = 1.f / lrun[r];
    int row = qrowbase + r + 8 * hi;
#pragma unroll
    for (int n = 0; n < 4; ++n)
      ctx16[(size_t)row * H_ + head * DH_ + n * 16 + lo] = (f16)(acc[n][r] * inv);
  }
}

// ---------------------------------------------------------------------------
// Final gather: out[b,j,:] = h[b, end[b,j], :] * indices_mask[b,j]
// ---------------------------------------------------------------------------
__global__ __launch_bounds__(256)
void gather_kernel(const float* __restrict__ h, const int* __restrict__ endi,
                   const float* __restrict__ imask, float* __restrict__ out) {
  int bj = blockIdx.x;
  int b  = bj / J_;
  int e  = endi[bj];
  float m = imask[bj];
  for (int c = threadIdx.x; c < H_; c += 256)
    out[(size_t)bj * H_ + c] = h[((size_t)b * S_ + e) * H_ + c] * m;
}

// ---------------------------------------------------------------------------
extern "C" void kernel_launch(void* const* d_in, const int* in_sizes, int n_in,
                              void* d_out, int out_size, void* d_ws, size_t ws_size,
                              hipStream_t stream) {
  (void)in_sizes; (void)n_in; (void)out_size; (void)ws_size;

  const int*   input_ids   = (const int*)  d_in[0];
  const int*   segment_ids = (const int*)  d_in[1];
  const float* padding     = (const float*)d_in[2];
  const int*   start_idx   = (const int*)  d_in[3];
  const int*   end_idx     = (const int*)  d_in[4];
  const float* idx_mask    = (const float*)d_in[5];
  const float* word_emb    = (const float*)d_in[6];
  const float* pos_emb     = (const float*)d_in[7];
  const float* type_emb    = (const float*)d_in[8];
  const float* emb_s       = (const float*)d_in[9];
  const float* emb_b       = (const float*)d_in[10];
  const float* Wq  = (const float*)d_in[11];
  const float* bq  = (const float*)d_in[12];
  const float* Wk  = (const float*)d_in[13];
  const float* bk  = (const float*)d_in[14];
  const float* Wv  = (const float*)d_in[15];
  const float* bv  = (const float*)d_in[16];
  const float* Wo  = (const float*)d_in[17];
  const float* bo  = (const float*)d_in[18];
  const float* ln1s = (const float*)d_in[19];
  const float* ln1b = (const float*)d_in[20];
  const float* Wi  = (const float*)d_in[21];
  const float* bi  = (const float*)d_in[22];
  const float* Wo2 = (const float*)d_in[23];
  const float* bo2 = (const float*)d_in[24];
  const float* ln2s = (const float*)d_in[25];
  const float* ln2b = (const float*)d_in[26];
  float* out = (float*)d_out;

  char* base = (char*)d_ws;
  size_t off = 0;
  auto take = [&](size_t bytes) -> void* {
    void* p = base + off;
    off += (bytes + 255) & ~(size_t)255;
    return p;
  };
  float* h     = (float*)take((size_t)TOK_ * H_ * 4);
  f16*   h16   = (f16*)  take((size_t)TOK_ * H_ * 2);
  f16*   q16   = (f16*)  take((size_t)TOK_ * H_ * 2);
  f16*   k16   = (f16*)  take((size_t)TOK_ * H_ * 2);
  f16*   v16   = (f16*)  take((size_t)TOK_ * H_ * 2);
  f16*   vt16  = (f16*)  take((size_t)TOK_ * H_ * 2);
  f16*   ctx16 = (f16*)  take((size_t)TOK_ * H_ * 2);
  f16*   ff16  = (f16*)  take((size_t)TOK_ * FF_ * 2);
  float* tmp   = (float*)take((size_t)TOK_ * H_ * 4);
  float* biasm = (float*)take((size_t)B_ * S_ * S_ * 4);
  f16*   wq_t  = (f16*)  take((size_t)H_ * H_ * 2);
  f16*   wk_t  = (f16*)  take((size_t)H_ * H_ * 2);
  f16*   wv_t  = (f16*)  take((size_t)H_ * H_ * 2);
  f16*   wo_t  = (f16*)  take((size_t)H_ * H_ * 2);
  f16*   wi_t  = (f16*)  take((size_t)H_ * FF_ * 2);
  f16*   wo2_t = (f16*)  take((size_t)FF_ * H_ * 2);

  embed_ln_kernel<<<TOK_, 32, 0, stream>>>(input_ids, segment_ids, word_emb,
                                           pos_emb, type_emb, emb_s, emb_b, h, h16);
  build_bias_kernel<<<dim3(S_, B_), 512, 0, stream>>>(start_idx, end_idx, padding, biasm);

  const dim3 gH(TOK_ / 128, H_ / 128);    // (32, 6)
  const dim3 gF(TOK_ / 128, FF_ / 128);   // (32, 24)
  const int nHH = (H_ * H_ + 255) / 256;
  const int nHF = (H_ * FF_ + 255) / 256;

  for (int l = 0; l < L_; ++l) {
    wtrans_kernel<<<nHH, 256, 0, stream>>>(Wq  + (size_t)l * H_ * H_,  wq_t,  H_,  H_);
    wtrans_kernel<<<nHH, 256, 0, stream>>>(Wk  + (size_t)l * H_ * H_,  wk_t,  H_,  H_);
    wtrans_kernel<<<nHH, 256, 0, stream>>>(Wv  + (size_t)l * H_ * H_,  wv_t,  H_,  H_);
    wtrans_kernel<<<nHH, 256, 0, stream>>>(Wo  + (size_t)l * H_ * H_,  wo_t,  H_,  H_);
    wtrans_kernel<<<nHF, 256, 0, stream>>>(Wi  + (size_t)l * H_ * FF_, wi_t,  H_,  FF_);
    wtrans_kernel<<<nHF, 256, 0, stream>>>(Wo2 + (size_t)l * FF_ * H_, wo2_t, FF_, H_);

    gemm_wmma_kernel<<<gH, 256, 0, stream>>>(h16, wq_t, bq + l * H_, nullptr, q16,
                                             TOK_, H_, H_, 0);
    gemm_wmma_kernel<<<gH, 256, 0, stream>>>(h16, wk_t, bk + l * H_, nullptr, k16,
                                             TOK_, H_, H_, 0);
    gemm_wmma_kernel<<<gH, 256, 0, stream>>>(h16, wv_t, bv + l * H_, nullptr, v16,
                                             TOK_, H_, H_, 0);
    vtrans_kernel<<<(TOK_ * H_ + 255) / 256, 256, 0, stream>>>(v16, vt16);

    attention_kernel<<<(B_ * NH_ * (S_ / 16)) / 4, 128, 0, stream>>>(q16, k16, vt16,
                                                                     biasm, ctx16);

    gemm_wmma_kernel<<<gH, 256, 0, stream>>>(ctx16, wo_t, bo + l * H_, tmp, nullptr,
                                             TOK_, H_, H_, 0);
    residual_ln_kernel<<<TOK_, 32, 0, stream>>>(h, tmp, ln1s + l * H_, ln1b + l * H_,
                                                h, h16);

    gemm_wmma_kernel<<<gF, 256, 0, stream>>>(h16, wi_t, bi + l * FF_, nullptr, ff16,
                                             TOK_, FF_, H_, 1);
    gemm_wmma_kernel<<<gH, 256, 0, stream>>>(ff16, wo2_t, bo2 + l * H_, tmp, nullptr,
                                             TOK_, H_, FF_, 0);
    residual_ln_kernel<<<TOK_, 32, 0, stream>>>(h, tmp, ln2s + l * H_, ln2b + l * H_,
                                                h, h16);
  }

  gather_kernel<<<B_ * J_, 256, 0, stream>>>(h, end_idx, idx_mask, out);
}
